// GroupedQueryAttention_41051297415403
// MI455X (gfx1250) — matrix operand order repaired and checked by reading, not compiled, not run on hardware
//
#include <hip/hip_runtime.h>

typedef __attribute__((ext_vector_type(16))) __bf16 v16bf;
typedef __attribute__((ext_vector_type(4)))  __bf16 bf16x4;
typedef __attribute__((ext_vector_type(8)))  float  v8f;

// ---------------------------------------------------------------------------
// WMMA helpers (CDNA5 gfx1250, wave32)
// ---------------------------------------------------------------------------
__device__ inline v8f wmma_bf16(v16bf a, v16bf b, v8f c) {
    // (neg_a, A, neg_b, B, c_mod, C, reuse_a, reuse_b)
    return __builtin_amdgcn_wmma_f32_16x16x32_bf16(false, a, false, b,
                                                   (short)0, c, false, false);
}

// A-matrix fragment (16x32 bf16, MxK): lane holds row m = lane%16.
// VGPR p holds K = (p/4)*16 + (lane/16)*8 + (p%4)*2 (+1).  [ISA 7.12.2]
__device__ inline v16bf frag_a_from_row(const __bf16* row, int hl) {
    union { unsigned int u[8]; v16bf v; } r;
#pragma unroll
    for (int p = 0; p < 8; ++p) {
        int kk = ((p >> 2) << 4) + (hl << 3) + ((p & 3) << 1);
        r.u[p] = *(const unsigned int*)(row + kk);
    }
    return r.v;
}

// B-matrix fragment (32x16 bf16, KxN): lane holds col n = lane%16.
// VGPR p holds K = (p/4)*16 + (lane/16)*8 + (p%4)*2 (+1), same as A.
__device__ inline v16bf frag_b_from_row(const __bf16* row, int hl) {
    union { unsigned int u[8]; v16bf v; } r;
#pragma unroll
    for (int p = 0; p < 8; ++p) {
        int kk = ((p >> 2) << 4) + (hl << 3) + ((p & 3) << 1);
        r.u[p] = *(const unsigned int*)(row + kk);
    }
    return r.v;
}

// CDNA5 async global->LDS DMA (per-lane, tracked by ASYNCcnt).
// lds byte offset = low 32 bits of the generic pointer (aperture rules).
__device__ inline void async_copy_b128(const __bf16* g, __bf16* l) {
    unsigned int lds = (unsigned int)(size_t)l;
    asm volatile("global_load_async_to_lds_b128 %0, %1, off"
                 :: "v"(lds), "v"(g) : "memory");
}
__device__ inline void async_wait0() {
    asm volatile("s_wait_asynccnt 0x0" ::: "memory");
}

// ---------------------------------------------------------------------------
// f32 -> bf16 cast
// ---------------------------------------------------------------------------
__global__ void f2bf_kernel(const float* __restrict__ in, __bf16* __restrict__ out, int n) {
    int i = blockIdx.x * 256 + threadIdx.x;
    if (i < n) out[i] = (__bf16)in[i];
}

// ---------------------------------------------------------------------------
// GEMM: out[M,N] = A[M,K] (bf16,row-major) * B[N,K]^T (bf16,row-major) + bias
// 128 threads (4 waves), 64x64 block tile, each wave owns a 32x32 quadrant.
// Tiles staged via global_load_async_to_lds_b128.
// ---------------------------------------------------------------------------
template <bool OUT_F32>
__global__ __launch_bounds__(128) void gemm_bf16_nt(
    const __bf16* __restrict__ A, const __bf16* __restrict__ B,
    const float* __restrict__ bias, void* __restrict__ out,
    int M, int N, int K) {
    constexpr int LDT = 40;  // 80B row stride: 16B aligned, bank-skewed
    __shared__ __bf16 As[64 * LDT];
    __shared__ __bf16 Bs[64 * LDT];

    const int tid  = threadIdx.x;
    const int lane = tid & 31, wave = tid >> 5;
    const int hl   = lane >> 4, l16 = lane & 15;
    const int m0 = blockIdx.y * 64, n0 = blockIdx.x * 64;
    const int mt = (wave >> 1) * 32, nt = (wave & 1) * 32;

    v8f acc[2][2] = {};

    for (int k0 = 0; k0 < K; k0 += 32) {
        __syncthreads();
        // 64x32 bf16 tile = 256 x 16B chunks; 2 chunks each for A and B
#pragma unroll
        for (int p = 0; p < 2; ++p) {
            int c = tid + p * 128;
            int row = c >> 2, cc = (c & 3) << 3;
            async_copy_b128(A + (size_t)(m0 + row) * K + k0 + cc, &As[row * LDT + cc]);
            async_copy_b128(B + (size_t)(n0 + row) * K + k0 + cc, &Bs[row * LDT + cc]);
        }
        async_wait0();
        __syncthreads();

        v16bf a0 = frag_a_from_row(&As[(mt + 0 + l16) * LDT], hl);
        v16bf a1 = frag_a_from_row(&As[(mt + 16 + l16) * LDT], hl);
#pragma unroll
        for (int j = 0; j < 2; ++j) {
            v16bf b = frag_b_from_row(&Bs[(nt + j * 16 + l16) * LDT], hl);
            acc[0][j] = wmma_bf16(a0, b, acc[0][j]);
            acc[1][j] = wmma_bf16(a1, b, acc[1][j]);
        }
    }

#pragma unroll
    for (int i = 0; i < 2; ++i)
#pragma unroll
        for (int j = 0; j < 2; ++j) {
            int col = n0 + nt + j * 16 + l16;
            float bs = bias[col];
#pragma unroll
            for (int r = 0; r < 8; ++r) {
                int row = m0 + mt + i * 16 + r + 8 * hl;
                float v = acc[i][j][r] + bs;
                if (OUT_F32)
                    ((float*)out)[(size_t)row * N + col] = v;
                else
                    ((__bf16*)out)[(size_t)row * N + col] = (__bf16)v;
            }
        }
}

// ---------------------------------------------------------------------------
// RoPE + head-major reorder
// Qp [b,t,2048] (channel = (h*4+g)*64+d)  ->  Qr [b,g,h,t,64]
// ---------------------------------------------------------------------------
__global__ void rope_q_kernel(const __bf16* __restrict__ Qp, __bf16* __restrict__ Qr) {
    int idx = blockIdx.x * 256 + threadIdx.x;   // b*t*1024 pairs
    int c  = idx & 1023;
    int bt = idx >> 10;
    int ti = bt & 2047, bi = bt >> 11;
    int hg = c >> 5, j = c & 31;
    int hi = hg >> 2, gi = hg & 3;
    float theta = __powf(10000.0f, -(float)j * (1.0f / 32.0f));
    float sn, cs;
    __sincosf((float)(ti + 1) * theta, &sn, &cs);
    size_t src = ((size_t)(bi * 2048 + ti)) * 2048 + hg * 64 + j;
    float x0 = (float)Qp[src];
    float x1 = (float)Qp[src + 32];
    size_t dst = (((size_t)((bi * 4 + gi) * 8 + hi)) * 2048 + ti) * 64 + j;
    Qr[dst]      = (__bf16)(x0 * cs - x1 * sn);
    Qr[dst + 32] = (__bf16)(x1 * cs + x0 * sn);
}

// Kp [b,t,512] (channel = h*64+d) -> Kr [b,h,t,64]
__global__ void rope_k_kernel(const __bf16* __restrict__ Kp, __bf16* __restrict__ Kr) {
    int idx = blockIdx.x * 256 + threadIdx.x;   // b*t*256 pairs
    int c  = idx & 255;
    int bt = idx >> 8;
    int ti = bt & 2047, bi = bt >> 11;
    int hi = c >> 5, j = c & 31;
    float theta = __powf(10000.0f, -(float)j * (1.0f / 32.0f));
    float sn, cs;
    __sincosf((float)(ti + 1) * theta, &sn, &cs);
    size_t src = ((size_t)(bi * 2048 + ti)) * 512 + hi * 64 + j;
    float x0 = (float)Kp[src];
    float x1 = (float)Kp[src + 32];
    size_t dst = (((size_t)(bi * 8 + hi)) * 2048 + ti) * 64 + j;
    Kr[dst]      = (__bf16)(x0 * cs - x1 * sn);
    Kr[dst + 32] = (__bf16)(x1 * cs + x0 * sn);
}

// Vp [b,t,512] -> Vr [b,h,t,64]
__global__ void reorder_v_kernel(const __bf16* __restrict__ Vp, __bf16* __restrict__ Vr) {
    int idx = blockIdx.x * 256 + threadIdx.x;   // b*t*512 elems
    int c  = idx & 511;
    int bt = idx >> 9;
    int ti = bt & 2047, bi = bt >> 11;
    int hi = c >> 6, di = c & 63;
    Vr[(((size_t)(bi * 8 + hi)) * 2048 + ti) * 64 + di] =
        Vp[((size_t)(bi * 2048 + ti)) * 512 + c];
}

// ---------------------------------------------------------------------------
// Flash attention, causal.  grid = (t/64, b*g*h), 128 threads (4 waves).
// Each wave owns 16 q rows; k-tiles of 64 keys staged in LDS (K via async
// DMA, V transposed by VALU).  Output Y [b, t, (h g d)] bf16.
// ---------------------------------------------------------------------------
__global__ __launch_bounds__(128) void attn_kernel(
    const __bf16* __restrict__ Qr, const __bf16* __restrict__ Kr,
    const __bf16* __restrict__ Vr, __bf16* __restrict__ Y) {
    constexpr int T = 2048, D = 64, LDT = 80;  // 160B rows: 16B aligned
    __shared__ __bf16 Ks[64 * LDT];   // [key][d]
    __shared__ __bf16 Vt[64 * LDT];   // [d][key]  (transposed)
    __shared__ __bf16 Ps[64 * LDT];   // probabilities, per-wave 16-row slices

    const int tid = threadIdx.x, lane = tid & 31, wave = tid >> 5;
    const int hl = lane >> 4, l16 = lane & 15;
    const int hd = blockIdx.y;
    const int bi = hd >> 5, rem = hd & 31, gi = rem >> 3, hi = rem & 7;
    const int q0 = blockIdx.x * 64;
    const int qrow_base = q0 + wave * 16;

    const __bf16* Qh = Qr + (size_t)((bi * 4 + gi) * 8 + hi) * T * D;
    const __bf16* Kh = Kr + (size_t)(bi * 8 + hi) * T * D;
    const __bf16* Vh = Vr + (size_t)(bi * 8 + hi) * T * D;

    // Q fragments for this wave's 16 rows (contraction d=64 -> two chunks)
    v16bf qf0, qf1;
    {
        const __bf16* qrow = Qh + (size_t)(qrow_base + l16) * D;
        qf0 = frag_a_from_row(qrow, hl);
        qf1 = frag_a_from_row(qrow + 32, hl);
    }

    float mrow[8], lrow[8];
#pragma unroll
    for (int r = 0; r < 8; ++r) { mrow[r] = -3.0e38f; lrow[r] = 0.0f; }
    v8f o[4] = {};

    const int nkt = blockIdx.x + 1;
    for (int kt = 0; kt < nkt; ++kt) {
        const int kb = kt * 64;
        __syncthreads();
        // K tile: 64x64 bf16 = 512 x 16B chunks via async DMA
#pragma unroll
        for (int p = 0; p < 4; ++p) {
            int c = tid + p * 128;
            int row = c >> 3, cc = (c & 7) << 3;
            async_copy_b128(Kh + (size_t)(kb + row) * D + cc, &Ks[row * LDT + cc]);
        }
        // V tile, transposed store
#pragma unroll
        for (int p = 0; p < 8; ++p) {
            int c = tid + p * 128;              // 1024 chunks of 4
            int row = c >> 4, cc = (c & 15) << 2;
            bf16x4 v = *(const bf16x4*)(Vh + (size_t)(kb + row) * D + cc);
            Vt[(cc + 0) * LDT + row] = v[0];
            Vt[(cc + 1) * LDT + row] = v[1];
            Vt[(cc + 2) * LDT + row] = v[2];
            Vt[(cc + 3) * LDT + row] = v[3];
        }
        async_wait0();
        __syncthreads();

        // S = Q K^T  (16 x 64 per wave)
        v8f s[4] = {};
#pragma unroll
        for (int kc = 0; kc < 2; ++kc) {
            v16bf a = (kc == 0) ? qf0 : qf1;
#pragma unroll
            for (int j = 0; j < 4; ++j) {
                v16bf b = frag_b_from_row(&Ks[(j * 16 + l16) * LDT] + kc * 32, hl);
                s[j] = wmma_bf16(a, b, s[j]);
            }
        }

        // online softmax (rows replicated across the 16 lanes of each half)
        const float scale = 0.125f;  // 1/sqrt(64)
#pragma unroll
        for (int r = 0; r < 8; ++r) {
            const int m  = r + 8 * hl;
            const int qg = qrow_base + m;
            float sv[4];
            float rmax = -3.0e38f;
#pragma unroll
            for (int j = 0; j < 4; ++j) {
                float v = s[j][r] * scale;
                if (kb + j * 16 + l16 > qg) v = -3.0e38f;  // causal mask
                sv[j] = v;
                rmax = fmaxf(rmax, v);
            }
#pragma unroll
            for (int msk = 1; msk < 16; msk <<= 1)
                rmax = fmaxf(rmax, __shfl_xor(rmax, msk, 32));
            const float mn  = fmaxf(mrow[r], rmax);
            const float fac = __expf(mrow[r] - mn);
            float rsum = 0.0f;
#pragma unroll
            for (int j = 0; j < 4; ++j) {
                float p = __expf(sv[j] - mn);
                rsum += p;
                Ps[(wave * 16 + m) * LDT + j * 16 + l16] = (__bf16)p;
            }
#pragma unroll
            for (int msk = 1; msk < 16; msk <<= 1)
                rsum += __shfl_xor(rsum, msk, 32);
            lrow[r] = lrow[r] * fac + rsum;
            mrow[r] = mn;
#pragma unroll
            for (int j = 0; j < 4; ++j) o[j][r] = o[j][r] * fac;
        }

        // O += P V   (contraction over 64 keys -> two chunks)
#pragma unroll
        for (int kc = 0; kc < 2; ++kc) {
            v16bf a = frag_a_from_row(&Ps[(wave * 16 + l16) * LDT] + kc * 32, hl);
#pragma unroll
            for (int j = 0; j < 4; ++j) {
                v16bf b = frag_b_from_row(&Vt[(j * 16 + l16) * LDT] + kc * 32, hl);
                o[j] = wmma_bf16(a, b, o[j]);
            }
        }
    }

    // write Y[b, q, h*256 + g*64 + d]
#pragma unroll
    for (int r = 0; r < 8; ++r) {
        float invl = 1.0f / lrow[r];
        int q = qrow_base + r + 8 * hl;
#pragma unroll
        for (int j = 0; j < 4; ++j) {
            int ch = hi * 256 + gi * 64 + j * 16 + l16;
            Y[((size_t)bi * T + q) * 2048 + ch] = (__bf16)(o[j][r] * invl);
        }
    }
}

// ---------------------------------------------------------------------------
// Host launcher
// ---------------------------------------------------------------------------
extern "C" void kernel_launch(void* const* d_in, const int* in_sizes, int n_in,
                              void* d_out, int out_size, void* d_ws, size_t ws_size,
                              hipStream_t stream) {
    (void)in_sizes; (void)n_in; (void)out_size; (void)ws_size;
    const float* x  = (const float*)d_in[0];
    const float* Wq = (const float*)d_in[1];
    const float* bq = (const float*)d_in[2];
    const float* Wk = (const float*)d_in[3];
    const float* bk = (const float*)d_in[4];
    const float* Wv = (const float*)d_in[5];
    const float* bv = (const float*)d_in[6];
    const float* Wo = (const float*)d_in[7];
    const float* bo = (const float*)d_in[8];
    float* out = (float*)d_out;

    const int B = 2, T = 2048, E = 2048, KV = 512, M = B * T;

    char* ws = (char*)d_ws;
    size_t off = 0;
    auto alloc = [&](size_t bytes) -> void* {
        void* p = ws + off;
        off += (bytes + 255) & ~(size_t)255;
        return p;
    };
    __bf16* xb  = (__bf16*)alloc((size_t)M * E * 2);
    __bf16* Wqb = (__bf16*)alloc((size_t)E * E * 2);
    __bf16* Wkb = (__bf16*)alloc((size_t)KV * E * 2);
    __bf16* Wvb = (__bf16*)alloc((size_t)KV * E * 2);
    __bf16* Wob = (__bf16*)alloc((size_t)E * E * 2);
    __bf16* Qp  = (__bf16*)alloc((size_t)M * E * 2);
    __bf16* Kp  = (__bf16*)alloc((size_t)M * KV * 2);
    __bf16* Vp  = (__bf16*)alloc((size_t)M * KV * 2);
    __bf16* Qrr = (__bf16*)alloc((size_t)M * E * 2);
    __bf16* Krr = (__bf16*)alloc((size_t)M * KV * 2);
    __bf16* Vrr = (__bf16*)alloc((size_t)M * KV * 2);
    __bf16* Yb  = (__bf16*)alloc((size_t)M * E * 2);

    // cast inputs to bf16
    f2bf_kernel<<<(M * E + 255) / 256, 256, 0, stream>>>(x, xb, M * E);
    f2bf_kernel<<<(E * E + 255) / 256, 256, 0, stream>>>(Wq, Wqb, E * E);
    f2bf_kernel<<<(KV * E + 255) / 256, 256, 0, stream>>>(Wk, Wkb, KV * E);
    f2bf_kernel<<<(KV * E + 255) / 256, 256, 0, stream>>>(Wv, Wvb, KV * E);
    f2bf_kernel<<<(E * E + 255) / 256, 256, 0, stream>>>(Wo, Wob, E * E);

    // projections
    gemm_bf16_nt<false><<<dim3(E / 64, M / 64), 128, 0, stream>>>(xb, Wqb, bq, Qp, M, E, E);
    gemm_bf16_nt<false><<<dim3(KV / 64, M / 64), 128, 0, stream>>>(xb, Wkb, bk, Kp, M, KV, E);
    gemm_bf16_nt<false><<<dim3(KV / 64, M / 64), 128, 0, stream>>>(xb, Wvb, bv, Vp, M, KV, E);

    // RoPE + head-major reorder
    rope_q_kernel<<<(B * T * 1024) / 256, 256, 0, stream>>>(Qp, Qrr);
    rope_k_kernel<<<(B * T * 256) / 256, 256, 0, stream>>>(Kp, Krr);
    reorder_v_kernel<<<(B * T * 512) / 256, 256, 0, stream>>>(Vp, Vrr);

    // flash attention over 64 heads
    attn_kernel<<<dim3(T / 64, B * 4 * 8), 128, 0, stream>>>(Qrr, Krr, Vrr, Yb);

    // output projection (fp32 out)
    gemm_bf16_nt<true><<<dim3(E / 64, M / 64), 128, 0, stream>>>(Yb, Wob, bo, out, M, E, E);
}
